// ExpressionEstimator_Attention_33268816675262
// MI455X (gfx1250) — matrix operand order repaired and checked by reading, not compiled, not run on hardware
//
#include <hip/hip_runtime.h>
#include <math.h>

typedef _Float16 v16h __attribute__((ext_vector_type(16)));
typedef _Float16 h8   __attribute__((ext_vector_type(8)));
typedef _Float16 h4   __attribute__((ext_vector_type(4)));
typedef float    v8f  __attribute__((ext_vector_type(8)));

// ------------------------------ geometry ------------------------------
#define T_DIM     9
#define SPB       32                 // samples per block
#define THREADS   256                // 8 wave32
#define NWAVES    (THREADS / 32)
#define STRIDE    136                // padded LDS row stride in halfs (272B -> 4-bank skew)
#define NROWS     (T_DIM * SPB)      // 288 rows = 18 tiles of 16
#define RTILES    (NROWS / 16)

// ---------------------- f16 weight workspace (halfs) -------------------
#define W_C1_OFF 0
#define W_C1_N   (32*192)
#define W_C2_OFF (W_C1_OFF + W_C1_N)
#define W_C2_N   (64*96)
#define W_C3_OFF (W_C2_OFF + W_C2_N)
#define W_C3_N   (128*192)
#define W_C4_OFF (W_C3_OFF + W_C3_N)
#define W_C4_N   (64*384)
#define W_F1_OFF (W_C4_OFF + W_C4_N)
#define W_F1_N   (128*64)
#define W_F2_OFF (W_F1_OFF + W_F1_N)
#define W_F2_N   (64*128)
#define W_F3_OFF (W_F2_OFF + W_F2_N)
#define W_F3_N   (32*64)
#define W_TOTAL  (W_F3_OFF + W_F3_N)  // 79872 halfs

// ------------------------ LDS constant-buffer map (floats) -------------
#define CB_B1   0
#define CB_B2   32
#define CB_B3   96
#define CB_B4   224
#define CB_FB1  288
#define CB_FB2  416
#define CB_FB3  480
#define CB_AW1  512
#define CB_AB1  2048
#define CB_AW2  2064
#define CB_AB2  2448
#define CB_AW3  2456
#define CB_AB3  2552
#define CB_AW4  2556
#define CB_AB4  2580
#define CB_AW5  2582
#define CB_AB5  2588
#define CB_LW   2589
#define CB_LB   2670
#define CB_TOTAL 2688

// ------------------------------ LDS layout -----------------------------
#define OFF_BUFA 0
#define SZ_BUF   (NROWS * STRIDE * 2)          // 78336 B
#define OFF_BUFB (OFF_BUFA + SZ_BUF)
#define OFF_WB0  (OFF_BUFB + SZ_BUF)           // ping-pong weight stages
#define SZ_WBUF  49152                         // max layer = 24576 halfs
#define OFF_WB1  (OFF_WB0 + SZ_WBUF)
#define OFF_RS   (OFF_WB1 + SZ_WBUF)
#define SZ_RS    (SPB * 32 * 10 * 4)           // rs [s][c][t] f32, t padded to 10
#define OFF_CBUF (OFF_RS + SZ_RS)
#define SZ_CBUF  (CB_TOTAL * 4)
#define OFF_IDS  (OFF_CBUF + SZ_CBUF)
#define SMEM_BYTES (OFF_IDS + 128)             // = 306816 B  (< 320KB WGP LDS)

// ================== weight preprocessing: f32 -> f16 im2col ============
__device__ __forceinline__ void conv_fill(_Float16* dst, const float* src,
                                          int Cout, int Cin, int j) {
    int K  = Cin * 3;
    int co = j / K;
    int kk = j % K;
    int tap = kk / Cin;          // K is tap-major: K = tap*Cin + ci
    int ci  = kk % Cin;
    dst[j] = (_Float16)src[(co * Cin + ci) * 3 + tap];
}

__global__ void prep_weights(const float* __restrict__ cw1, const float* __restrict__ cw2,
                             const float* __restrict__ cw3, const float* __restrict__ cw4,
                             const float* __restrict__ fw1, const float* __restrict__ fw2,
                             const float* __restrict__ fw3, _Float16* __restrict__ ws) {
    for (int i = blockIdx.x * blockDim.x + threadIdx.x; i < W_TOTAL;
         i += gridDim.x * blockDim.x) {
        if      (i < W_C2_OFF) conv_fill(ws + W_C1_OFF, cw1,  32,  64, i - W_C1_OFF);
        else if (i < W_C3_OFF) conv_fill(ws + W_C2_OFF, cw2,  64,  32, i - W_C2_OFF);
        else if (i < W_C4_OFF) conv_fill(ws + W_C3_OFF, cw3, 128,  64, i - W_C3_OFF);
        else if (i < W_F1_OFF) conv_fill(ws + W_C4_OFF, cw4,  64, 128, i - W_C4_OFF);
        else if (i < W_F2_OFF) ws[i] = (_Float16)fw1[i - W_F1_OFF];
        else if (i < W_F3_OFF) ws[i] = (_Float16)fw2[i - W_F2_OFF];
        else                   ws[i] = (_Float16)fw3[i - W_F3_OFF];
    }
}

// ======================= WMMA fragment loaders =========================
// A (16x32 f16): lane<16 -> M=lane, K = k0..k0+15;
//                lane>=16 -> M=lane-16, K = k0+16..k0+31.
__device__ __forceinline__ v16h load_a_frag(const _Float16* W, int K, int mt,
                                            int kt, int lane) {
    int m = mt * 16 + (lane & 15);
    int b0 = (lane < 16) ? 0 : 16;
    const _Float16* p = W + m * K + kt * 32 + b0;
    union { v16h v; h8 h[2]; } u;
    u.h[0] = *(const h8*)(p);
    u.h[1] = *(const h8*)(p + 8);
    return u.v;
}

// B (32x16 f16): lane<16 -> N=lane, halfs = K k0..k0+15;
//                lane>=16 -> N=lane-16, halfs = K k0+16..k0+31.
__device__ __forceinline__ v16h load_b_frag_fc(const _Float16* X, int rt, int kt,
                                               int lane) {
    int r = rt * 16 + (lane & 15);
    int k = kt * 32 + ((lane < 16) ? 0 : 16);
    const _Float16* p = X + r * STRIDE + k;
    union { v16h v; h8 h[2]; } u;
    u.h[0] = *(const h8*)(p);
    u.h[1] = *(const h8*)(p + 8);
    return u.v;
}

// Conv im2col B-fragment: K = tap*Cin + ci; each 32-wide k-tile stays inside
// one tap (Cin in {32,64,128}); t-shift is uniform per 16-row tile.
__device__ __forceinline__ v16h load_b_frag_conv(const _Float16* X, int Cin,
                                                 int rt, int kt, int lane) {
    int r  = rt * 16 + (lane & 15);
    int t  = r >> 5;            // row = t*32 + s
    int s  = r & 31;
    int kk = kt * 32;
    int tap = kk / Cin;
    int ci  = (kk % Cin) + ((lane < 16) ? 0 : 16);
    int tp  = t + tap - 1;      // SAME padding
    union { v16h v; h8 h[2]; } u;
    if (tp >= 0 && tp < T_DIM) {
        const _Float16* p = X + (tp * 32 + s) * STRIDE + ci;
        u.h[0] = *(const h8*)(p);
        u.h[1] = *(const h8*)(p + 8);
    } else {
#pragma unroll
        for (int i = 0; i < 16; ++i) u.v[i] = (_Float16)0.0f;
    }
    return u.v;
}

// ==================== templated WMMA GEMM stage ========================
// Y[cout,row] = act(W . X + bias).  Row tiles processed in pairs sharing one
// A fragment (both tiles of a pair live in the same t -> same im2col shift);
// two independent accumulators cover the WMMA->WMMA hazard window.
// MODE 0: leakyReLU -> f16 act buffer;  MODE 1 (fc3): tanh -> f32 rs[s][c][t].
template <int COUT, int K, int CIN, int MODE>
__device__ __forceinline__ void gemm_stage(const _Float16* __restrict__ Xs,
                                           _Float16* __restrict__ Ys,
                                           float* __restrict__ rsOut,
                                           const _Float16* __restrict__ Ws,
                                           const float* __restrict__ bias, int tid) {
    const int lane = tid & 31;
    const int wave = __builtin_amdgcn_readfirstlane(tid >> 5);  // scalar loop, EXEC stays full
    constexpr int MT = COUT / 16;
    constexpr int RP = RTILES / 2;   // 9 row-pairs
    constexpr int NT = MT * RP;
    constexpr int KT = K / 32;
    for (int tile = wave; tile < NT; tile += NWAVES) {
        const int mt  = tile / RP;
        const int rp  = tile % RP;
        const int rt0 = 2 * rp, rt1 = 2 * rp + 1;
        v8f acc0, acc1;
#pragma unroll
        for (int i = 0; i < 8; ++i) { acc0[i] = 0.0f; acc1[i] = 0.0f; }
#pragma unroll
        for (int kt = 0; kt < KT; ++kt) {
            v16h a = load_a_frag(Ws, K, mt, kt, lane);
            v16h b0, b1;
            if (CIN > 0) {
                b0 = load_b_frag_conv(Xs, CIN, rt0, kt, lane);
                b1 = load_b_frag_conv(Xs, CIN, rt1, kt, lane);
            } else {
                b0 = load_b_frag_fc(Xs, rt0, kt, lane);
                b1 = load_b_frag_fc(Xs, rt1, kt, lane);
            }
            acc0 = __builtin_amdgcn_wmma_f32_16x16x32_f16(
                false, a, false, b0, (short)0, acc0, false, false);
            acc1 = __builtin_amdgcn_wmma_f32_16x16x32_f16(
                false, a, false, b1, (short)0, acc1, false, false);
        }
        // C/D layout: lane<16 -> N=lane, VGPR i -> M=i ; lane>=16 -> M=8+i.
        const int n = lane & 15;
        const int mBase = mt * 16 + ((lane < 16) ? 0 : 8);
#pragma unroll
        for (int hh = 0; hh < 2; ++hh) {
            v8f acc = hh ? acc1 : acc0;
            int r = (hh ? rt1 : rt0) * 16 + n;
            if (MODE == 1) {
                int t = r >> 5, s = r & 31;
#pragma unroll
                for (int i = 0; i < 8; ++i) {
                    int c = mBase + i;
                    rsOut[(s * 32 + c) * 10 + t] = tanhf(acc[i] + bias[c]);
                }
            } else {
                h8 outv;
#pragma unroll
                for (int i = 0; i < 8; ++i) {
                    float v = acc[i] + bias[mBase + i];
                    v = (v >= 0.0f) ? v : 0.02f * v;
                    outv[i] = (_Float16)v;
                }
                *(h8*)(Ys + r * STRIDE + mBase) = outv;
            }
        }
    }
}

// ------------------------------ helpers --------------------------------
__device__ __forceinline__ void cpf(float* dst, const float* src, int n, int tid) {
    for (int i = tid; i < n; i += THREADS) dst[i] = src[i];
}
__device__ __forceinline__ void load_w(_Float16* dst, const _Float16* src, int n,
                                       int tid) {
    const h8* s = (const h8*)src;
    h8* d = (h8*)dst;
    for (int i = tid; i < (n >> 3); i += THREADS) d[i] = s[i];
}
// tiny attention conv: layout [s][c][t], per-channel stride 10, f32 VALU
__device__ __forceinline__ void attn_conv(const float* in, float* out,
                                          const float* w, const float* b,
                                          int Cin, int Cout, int tid) {
    int total = SPB * Cout * T_DIM;
    for (int idx = tid; idx < total; idx += THREADS) {
        int s = idx / (Cout * T_DIM);
        int rem = idx % (Cout * T_DIM);
        int co = rem / T_DIM;
        int t = rem % T_DIM;
        float acc = b[co];
        for (int ci = 0; ci < Cin; ++ci) {
            const float* ip = in + (s * Cin + ci) * 10;
            const float* wp = w + (co * Cin + ci) * 3;
            if (t > 0) acc += wp[0] * ip[t - 1];
            acc += wp[1] * ip[t];
            if (t < T_DIM - 1) acc += wp[2] * ip[t + 1];
        }
        out[(s * Cout + co) * 10 + t] = (acc >= 0.0f) ? acc : 0.02f * acc;
    }
}

// ============================ fused kernel =============================
__global__ __launch_bounds__(THREADS)
void fused_expr_attn(const float* __restrict__ x, const int* __restrict__ ids,
                     const _Float16* __restrict__ wsh,
                     const float* __restrict__ cb1, const float* __restrict__ cb2,
                     const float* __restrict__ cb3, const float* __restrict__ cb4,
                     const float* __restrict__ fb1, const float* __restrict__ fb2,
                     const float* __restrict__ fb3,
                     const float* __restrict__ aw1, const float* __restrict__ ab1,
                     const float* __restrict__ aw2, const float* __restrict__ ab2,
                     const float* __restrict__ aw3, const float* __restrict__ ab3,
                     const float* __restrict__ aw4, const float* __restrict__ ab4,
                     const float* __restrict__ aw5, const float* __restrict__ ab5,
                     const float* __restrict__ lwm, const float* __restrict__ lbm,
                     const float* __restrict__ mapping, float* __restrict__ out) {
    extern __shared__ char smemRaw[];
    _Float16* bufA = (_Float16*)(smemRaw + OFF_BUFA);
    _Float16* bufB = (_Float16*)(smemRaw + OFF_BUFB);
    _Float16* wb0  = (_Float16*)(smemRaw + OFF_WB0);
    _Float16* wb1  = (_Float16*)(smemRaw + OFF_WB1);
    float*    rsb  = (float*)(smemRaw + OFF_RS);
    float*    cbuf = (float*)(smemRaw + OFF_CBUF);
    int*      sid  = (int*)(smemRaw + OFF_IDS);
    float*    attA = (float*)(smemRaw + OFF_WB0);   // alias wb0 after GEMMs
    float*    attB = attA + SPB * 16 * 10;          // 5120 floats each

    const int tid = threadIdx.x;
    const long b0 = (long)blockIdx.x * SPB;

    // ---- phase 0: constants / ids / x -> bufA f16 [(t,s)][c] / conv1 W ----
    cpf(cbuf + CB_B1,  cb1,  32, tid);  cpf(cbuf + CB_B2,  cb2,  64, tid);
    cpf(cbuf + CB_B3,  cb3, 128, tid);  cpf(cbuf + CB_B4,  cb4,  64, tid);
    cpf(cbuf + CB_FB1, fb1, 128, tid);  cpf(cbuf + CB_FB2, fb2,  64, tid);
    cpf(cbuf + CB_FB3, fb3,  32, tid);
    cpf(cbuf + CB_AW1, aw1, 1536, tid); cpf(cbuf + CB_AB1, ab1, 16, tid);
    cpf(cbuf + CB_AW2, aw2,  384, tid); cpf(cbuf + CB_AB2, ab2,  8, tid);
    cpf(cbuf + CB_AW3, aw3,   96, tid); cpf(cbuf + CB_AB3, ab3,  4, tid);
    cpf(cbuf + CB_AW4, aw4,   24, tid); cpf(cbuf + CB_AB4, ab4,  2, tid);
    cpf(cbuf + CB_AW5, aw5,    6, tid); cpf(cbuf + CB_AB5, ab5,  1, tid);
    cpf(cbuf + CB_LW,  lwm,   81, tid); cpf(cbuf + CB_LB,  lbm,  9, tid);
    if (tid < SPB) sid[tid] = ids[b0 + tid];
    for (int i = tid; i < SPB * T_DIM * 16; i += THREADS) {   // 4608 float4s
        int c4 = i & 15;
        int t  = (i >> 4) % T_DIM;
        int s  = i / (16 * T_DIM);
        float4 v = *(const float4*)(x + ((b0 + s) * T_DIM + t) * 64 + c4 * 4);
        h4 hv;
        hv[0] = (_Float16)v.x; hv[1] = (_Float16)v.y;
        hv[2] = (_Float16)v.z; hv[3] = (_Float16)v.w;
        *(h4*)(bufA + (t * 32 + s) * STRIDE + c4 * 4) = hv;
    }
    load_w(wb0, wsh + W_C1_OFF, W_C1_N, tid);
    __syncthreads();

    // warm L2 for the identity-gather epilogue (global_prefetch_b8)
    for (int i = tid; i < SPB * 53; i += THREADS) {
        int s = i / 53, e = i % 53;
        __builtin_prefetch((const char*)(mapping + ((size_t)sid[s] * 53 + e) * 32), 0, 1);
    }

    // ---- conv stack (WMMA); next layer's weights load in the same phase ---
    gemm_stage< 32, 192,  64, 0>(bufA, bufB, nullptr, wb0, cbuf + CB_B1, tid);
    load_w(wb1, wsh + W_C2_OFF, W_C2_N, tid);
    __syncthreads();
    gemm_stage< 64,  96,  32, 0>(bufB, bufA, nullptr, wb1, cbuf + CB_B2, tid);
    load_w(wb0, wsh + W_C3_OFF, W_C3_N, tid);
    __syncthreads();
    gemm_stage<128, 192,  64, 0>(bufA, bufB, nullptr, wb0, cbuf + CB_B3, tid);
    load_w(wb1, wsh + W_C4_OFF, W_C4_N, tid);
    __syncthreads();
    gemm_stage< 64, 384, 128, 0>(bufB, bufA, nullptr, wb1, cbuf + CB_B4, tid);
    load_w(wb0, wsh + W_F1_OFF, W_F1_N, tid);
    __syncthreads();

    // torch reshape quirk: flat (64ch,9t) -> 9 FC rows of 64 (mixes C and T)
    for (int i = tid; i < SPB * 576; i += THREADS) {
        int s = i / 576, f = i % 576;
        int ri = f >> 6, j = f & 63;          // FC row / col
        int c = f / 9, t = f % 9;             // conv channel / time
        bufB[(ri * 32 + s) * STRIDE + j] = bufA[(t * 32 + s) * STRIDE + c];
    }
    __syncthreads();

    // --------------------------- FC stack (WMMA) --------------------------
    gemm_stage<128,  64, 0, 0>(bufB, bufA, nullptr, wb0, cbuf + CB_FB1, tid);
    load_w(wb1, wsh + W_F2_OFF, W_F2_N, tid);
    __syncthreads();
    gemm_stage< 64, 128, 0, 0>(bufA, bufB, nullptr, wb1, cbuf + CB_FB2, tid);
    load_w(wb0, wsh + W_F3_OFF, W_F3_N, tid);
    __syncthreads();
    gemm_stage< 32,  64, 0, 1>(bufB, nullptr, rsb, wb0, cbuf + CB_FB3, tid);
    __syncthreads();

    // ----------------------- attention chain (VALU f32) -------------------
    attn_conv(rsb,  attA, cbuf + CB_AW1, cbuf + CB_AB1, 32, 16, tid); __syncthreads();
    attn_conv(attA, attB, cbuf + CB_AW2, cbuf + CB_AB2, 16,  8, tid); __syncthreads();
    attn_conv(attB, attA, cbuf + CB_AW3, cbuf + CB_AB3,  8,  4, tid); __syncthreads();
    attn_conv(attA, attB, cbuf + CB_AW4, cbuf + CB_AB4,  4,  2, tid); __syncthreads();
    attn_conv(attB, attA, cbuf + CB_AW5, cbuf + CB_AB5,  2,  1, tid); __syncthreads();

    // logits (9x9) + softmax, one thread per sample; att -> attB[s*10+t]
    if (tid < SPB) {
        int s = tid;
        float a5[9], lg[9];
#pragma unroll
        for (int t = 0; t < 9; ++t) a5[t] = attA[s * 10 + t];
        float mx = -1e30f;
#pragma unroll
        for (int t = 0; t < 9; ++t) {
            float v = cbuf[CB_LB + t];
#pragma unroll
            for (int u = 0; u < 9; ++u) v += a5[u] * cbuf[CB_LW + t * 9 + u];
            lg[t] = v;
            mx = fmaxf(mx, v);
        }
        float sum = 0.0f;
#pragma unroll
        for (int t = 0; t < 9; ++t) { lg[t] = expf(lg[t] - mx); sum += lg[t]; }
        float inv = 1.0f / sum;
#pragma unroll
        for (int t = 0; t < 9; ++t) attB[s * 10 + t] = lg[t] * inv;
    }
    __syncthreads();

    // sub[s][c] = sum_t rs[s][c][t] * att[s][t]   (attA region reused)
    float* subb = attA;
    for (int i = tid; i < SPB * 32; i += THREADS) {
        int s = i >> 5, c = i & 31;
        float acc = 0.0f;
#pragma unroll
        for (int t = 0; t < 9; ++t) acc += rsb[(s * 32 + c) * 10 + t] * attB[s * 10 + t];
        subb[i] = acc;
    }
    __syncthreads();

    // out[b][e] = 10 * sum_c mapping[id[b]][e][c] * sub[b][c]  (L2-resident)
    for (int i = tid; i < SPB * 53; i += THREADS) {
        int s = i / 53, e = i % 53;
        const float4* m4 = (const float4*)(mapping + ((size_t)sid[s] * 53 + e) * 32);
        float acc = 0.0f;
#pragma unroll
        for (int q = 0; q < 8; ++q) {
            float4 mv = m4[q];
            acc += mv.x * subb[s * 32 + q * 4 + 0] + mv.y * subb[s * 32 + q * 4 + 1]
                 + mv.z * subb[s * 32 + q * 4 + 2] + mv.w * subb[s * 32 + q * 4 + 3];
        }
        out[(b0 + s) * 53 + e] = 10.0f * acc;
    }
}

// ============================== launcher ===============================
extern "C" void kernel_launch(void* const* d_in, const int* in_sizes, int n_in,
                              void* d_out, int out_size, void* d_ws, size_t ws_size,
                              hipStream_t stream) {
    const float* x   = (const float*)d_in[0];
    const int*   ids = (const int*)d_in[1];
    const float* cw1 = (const float*)d_in[2];  const float* cb1 = (const float*)d_in[3];
    const float* cw2 = (const float*)d_in[4];  const float* cb2 = (const float*)d_in[5];
    const float* cw3 = (const float*)d_in[6];  const float* cb3 = (const float*)d_in[7];
    const float* cw4 = (const float*)d_in[8];  const float* cb4 = (const float*)d_in[9];
    const float* fw1 = (const float*)d_in[10]; const float* fb1 = (const float*)d_in[11];
    const float* fw2 = (const float*)d_in[12]; const float* fb2 = (const float*)d_in[13];
    const float* fw3 = (const float*)d_in[14]; const float* fb3 = (const float*)d_in[15];
    const float* aw1 = (const float*)d_in[16]; const float* ab1 = (const float*)d_in[17];
    const float* aw2 = (const float*)d_in[18]; const float* ab2 = (const float*)d_in[19];
    const float* aw3 = (const float*)d_in[20]; const float* ab3 = (const float*)d_in[21];
    const float* aw4 = (const float*)d_in[22]; const float* ab4 = (const float*)d_in[23];
    const float* aw5 = (const float*)d_in[24]; const float* ab5 = (const float*)d_in[25];
    const float* lwm = (const float*)d_in[26]; const float* lbm = (const float*)d_in[27];
    const float* map = (const float*)d_in[28];
    float* out = (float*)d_out;
    _Float16* wsh = (_Float16*)d_ws;

    prep_weights<<<(W_TOTAL + 255) / 256, 256, 0, stream>>>(cw1, cw2, cw3, cw4,
                                                            fw1, fw2, fw3, wsh);
    const int B = in_sizes[1];                 // 32768 samples
    fused_expr_attn<<<B / SPB, THREADS, SMEM_BYTES, stream>>>(
        x, ids, wsh, cb1, cb2, cb3, cb4, fb1, fb2, fb3,
        aw1, ab1, aw2, ab2, aw3, ab3, aw4, ab4, aw5, ab5,
        lwm, lbm, map, out);
}
